// Edge_Sheaf_NNet_24464133718392
// MI455X (gfx1250) — matrix operand order repaired and checked by reading, not compiled, not run on hardware
//
#include <hip/hip_runtime.h>
#include <hip/hip_bf16.h>
#include <math.h>

// ---------------------------------------------------------------------------
// Types for CDNA5 WMMA (wave32, 16x16x32 f16 -> f32)
// ---------------------------------------------------------------------------
typedef __attribute__((ext_vector_type(16))) _Float16 v16h;
typedef __attribute__((ext_vector_type(8)))  _Float16 v8h;
typedef __attribute__((ext_vector_type(8)))  float    v8f;

__device__ __forceinline__ v8f wmma16(v16h a, v16h b, v8f c) {
    return __builtin_amdgcn_wmma_f32_16x16x32_f16(
        /*neg_a=*/false, a, /*neg_b=*/false, b,
        /*c_mod=*/(short)0, c, /*reuse_a=*/false, /*reuse_b=*/false);
}

// A fragment (16x32, M=rows, K=cols) from a row-major f16 buffer.
// Lane layout: lanes 0-15 row=lane, elems 0..15 = K0..15;
// lanes 16-31 same rows, K16..31.
__device__ __forceinline__ v16h load_a_frag(const _Float16* base, int rowStride,
                                            int lane, int kt) {
    const int row  = lane & 15;
    const int koff = (lane & 16) ? 16 : 0;
    const _Float16* p = base + row * rowStride + kt * 32 + koff;
    v8h lo = *(const v8h*)p;
    v8h hi = *(const v8h*)(p + 8);
    return __builtin_shufflevector(lo, hi, 0,1,2,3,4,5,6,7,8,9,10,11,12,13,14,15);
}

// B fragment from pre-packed fragment storage: 32 lanes x 16 contiguous halfs.
__device__ __forceinline__ v16h load_b_frag(const _Float16* fragBase, int lane) {
    const _Float16* p = fragBase + lane * 16;
    v8h lo = *(const v8h*)p;
    v8h hi = *(const v8h*)(p + 8);
    return __builtin_shufflevector(lo, hi, 0,1,2,3,4,5,6,7,8,9,10,11,12,13,14,15);
}

// B fragment (32x16, K x N) from a matrix stored K-contiguous per column
// (i.e. transposed row-major: element [K,N] at N*64 + K). Two 16B loads.
__device__ __forceinline__ v16h load_b_frag_kc(const _Float16* baseT, int lane, int kt,
                                               int ncol0) {
    const int n  = ncol0 + (lane & 15);
    const int kb = kt * 32 + ((lane & 16) ? 16 : 0);
    const _Float16* p = baseT + n * 64 + kb;
    v8h lo = *(const v8h*)p;
    v8h hi = *(const v8h*)(p + 8);
    return __builtin_shufflevector(lo, hi, 0,1,2,3,4,5,6,7,8,9,10,11,12,13,14,15);
}

// ---------------------------------------------------------------------------
// Constants
// ---------------------------------------------------------------------------
#define DIMX   64
#define INDIM  128          // 2*DIMX MLP input
#define NHID   128          // hidden width
#define OUTD   4096         // DIMX*DIMX
#define HIDL   7            // hidden layers with ReLU
#define EPB    32           // edges per block (two 16-row WMMA tiles)
// packed f16 weight layout: layers 0..6 each 4 ktiles * 8 ntiles * 512 = 16384 halfs,
// layer 7: 4 ktiles * 256 ntiles * 512 = 524288 halfs
#define W7_OFF   (7 * 16384)
#define WP_TOTAL (W7_OFF + 4 * 256 * 512)
#define B7_OFF   (7 * 128)
#define BB_TOTAL (B7_OFF + 4096)

// ---------------------------------------------------------------------------
// Prep: pack fp32 weights into WMMA B-fragment order (f16)
// ---------------------------------------------------------------------------
__global__ void __launch_bounds__(256) pack_w_kernel(
    const float* __restrict__ W0, const float* __restrict__ W1,
    const float* __restrict__ W2, const float* __restrict__ W3,
    const float* __restrict__ W4, const float* __restrict__ W5,
    const float* __restrict__ W6, const float* __restrict__ W7,
    _Float16* __restrict__ out)
{
    int idx = blockIdx.x * 256 + threadIdx.x;
    if (idx >= WP_TOTAL) return;
    const float* const Ws[8] = {W0, W1, W2, W3, W4, W5, W6, W7};
    const float* W;
    int rem, N;
    if (idx < W7_OFF) { W = Ws[idx >> 14]; rem = idx & 16383; N = NHID; }
    else              { W = W7; rem = idx - W7_OFF; N = OUTD; }
    int frag  = rem >> 9;
    int r     = rem & 511;
    int lane  = r >> 4;
    int e     = r & 15;
    int ntile = frag >> 2;
    int kt    = frag & 3;
    int k = kt * 32 + ((lane & 16) ? 16 : 0) + e;
    int n = ntile * 16 + (lane & 15);
    out[idx] = (_Float16)W[(size_t)k * N + n];
}

__global__ void __launch_bounds__(256) pack_b_kernel(
    const float* __restrict__ b0, const float* __restrict__ b1,
    const float* __restrict__ b2, const float* __restrict__ b3,
    const float* __restrict__ b4, const float* __restrict__ b5,
    const float* __restrict__ b6, const float* __restrict__ b7,
    float* __restrict__ out)
{
    int idx = blockIdx.x * 256 + threadIdx.x;
    if (idx >= BB_TOTAL) return;
    const float* const bs[7] = {b0, b1, b2, b3, b4, b5, b6};
    if (idx < B7_OFF) out[idx] = bs[idx >> 7][idx & 127];
    else              out[idx] = b7[idx - B7_OFF];
}

// cnt[v] = segment_sum(edge_weight, edge_dst)  (iteration-invariant)
__global__ void __launch_bounds__(256) cnt_kernel(
    const int* __restrict__ edst, const float* __restrict__ ew,
    float* __restrict__ cnt, int E)
{
    int e = blockIdx.x * 256 + threadIdx.x;
    if (e < E) atomicAdd(&cnt[edst[e]], ew[e]);
}

// ---------------------------------------------------------------------------
// Edge MLP kernel. 32 edges / block (two 16-row tiles), 256 threads = 8 waves.
// Hidden layers: wave w owns N-tile w; 8 WMMAs/layer (2 tiles share each B frag).
// Final layer: wave w owns N-tiles 32w..32w+31; 256 WMMAs, each B frag reused 2x.
// STORE mode: write M f16; edges >= EH stored TRANSPOSED (K-contiguous) so the
//             orth kernel's B fragments are contiguous loads.
// FUSED mode: fold matvec msgs[e,i]=sum_j M[e,i,j]*x[src,j] into the epilogue,
//             scatter-add w[e]*msgs into num[dst] (M never materialized).
// ---------------------------------------------------------------------------
template <bool FUSED>
__global__ void __launch_bounds__(256) mlp_edge_kernel(
    const float* __restrict__ x,      // [NV,64] current vertex features
    const int*   __restrict__ esrc,
    const int*   __restrict__ edst,
    const float* __restrict__ ew,
    const _Float16* __restrict__ Wp,  // packed weights
    const float* __restrict__ Bb,     // packed biases
    _Float16* __restrict__ Mout,      // STORE: [E,4096] f16
    float* __restrict__ num,          // FUSED: [NV,64] f32 (atomic accum)
    int EHn)
{
    __shared__ __align__(16) _Float16 act[2][EPB][INDIM];
    __shared__ __align__(16) float xsrcS[EPB][DIMX];
    __shared__ __align__(16) float lmsg[EPB][DIMX];
    __shared__ int   edstS[EPB];
    __shared__ float ewS[EPB];

    const int tid   = threadIdx.x;
    const int lane  = tid & 31;
    const int wave  = tid >> 5;
    const int e0    = blockIdx.x * EPB;
    const int col   = lane & 15;
    const int rbase = (lane & 16) ? 8 : 0;

    // Stage inputs: act[0][i][0..63] = f16(x[src]), [64..127] = f16(x[dst])
    {
        int i    = tid >> 3;              // edge 0..31
        int cblk = (tid & 7) * 16;        // 0,16,...,112 (never straddles 64)
        int v    = (cblk < 64) ? esrc[e0 + i] : edst[e0 + i];
        int c    = cblk & 63;
        const float4* pf = (const float4*)(x + (size_t)v * DIMX + c);
        _Float16* d = &act[0][i][cblk];
        #pragma unroll
        for (int q = 0; q < 4; ++q) {
            float4 f = pf[q];
            d[q*4+0] = (_Float16)f.x; d[q*4+1] = (_Float16)f.y;
            d[q*4+2] = (_Float16)f.z; d[q*4+3] = (_Float16)f.w;
        }
        if (FUSED) {
            int c0 = (tid & 7) * 8;
            int s  = esrc[e0 + i];
            *(float4*)&xsrcS[i][c0]     = *(const float4*)(x + (size_t)s * DIMX + c0);
            *(float4*)&xsrcS[i][c0 + 4] = *(const float4*)(x + (size_t)s * DIMX + c0 + 4);
            *(float4*)&lmsg[i][c0]      = make_float4(0.f, 0.f, 0.f, 0.f);
            *(float4*)&lmsg[i][c0 + 4]  = make_float4(0.f, 0.f, 0.f, 0.f);
            if (tid < EPB) { edstS[tid] = edst[e0 + tid]; ewS[tid] = ew[e0 + tid]; }
        }
    }
    __syncthreads();

    // 7 hidden layers (128->128, ReLU), double-buffered in LDS
    int cur = 0;
    for (int l = 0; l < HIDL; ++l) {
        v8f c0a = {}, c1a = {};
        const _Float16* Wl = Wp + l * 16384;
        #pragma unroll
        for (int kt = 0; kt < 4; ++kt) {
            v16h b  = load_b_frag(Wl + (wave * 4 + kt) * 512, lane);
            v16h a0 = load_a_frag(&act[cur][0][0],  INDIM, lane, kt);
            v16h a1 = load_a_frag(&act[cur][16][0], INDIM, lane, kt);
            c0a = wmma16(a0, b, c0a);
            c1a = wmma16(a1, b, c1a);
        }
        float bv = Bb[l * 128 + wave * 16 + col];
        #pragma unroll
        for (int r = 0; r < 8; ++r) {
            float v0 = c0a[r] + bv, v1 = c1a[r] + bv;
            act[cur ^ 1][rbase + r][wave * 16 + col]      = (_Float16)(v0 > 0.f ? v0 : 0.f);
            act[cur ^ 1][16 + rbase + r][wave * 16 + col] = (_Float16)(v1 > 0.f ? v1 : 0.f);
        }
        __syncthreads();
        cur ^= 1;
    }

    // Final layer 128 -> 4096 (no ReLU); A fragments cached in registers
    v16h a4[2][4];
    #pragma unroll
    for (int kt = 0; kt < 4; ++kt) {
        a4[0][kt] = load_a_frag(&act[cur][0][0],  INDIM, lane, kt);
        a4[1][kt] = load_a_frag(&act[cur][16][0], INDIM, lane, kt);
    }
    const _Float16* W7 = Wp + W7_OFF;
    const float*    b7 = Bb + B7_OFF;

    if (!FUSED) {
        const bool second = (e0 >= EHn);   // B-operand half: store transposed
        for (int t = 0; t < 32; ++t) {
            int ntile = wave * 32 + t;
            if (t < 31) {                  // prefetch next iteration's B fragments
                #pragma unroll
                for (int kt = 0; kt < 4; ++kt)
                    __builtin_prefetch(W7 + ((ntile + 1) * 4 + kt) * 512 + lane * 16, 0, 3);
            }
            v8f c0a = {}, c1a = {};
            #pragma unroll
            for (int kt = 0; kt < 4; ++kt) {
                v16h b = load_b_frag(W7 + (ntile * 4 + kt) * 512, lane);
                c0a = wmma16(a4[0][kt], b, c0a);
                c1a = wmma16(a4[1][kt], b, c1a);
            }
            float bv = b7[ntile * 16 + col];
            int   n  = ntile * 16 + col;
            size_t off = second ? (size_t)((n & 63) * 64 + (n >> 6)) : (size_t)n;
            #pragma unroll
            for (int r = 0; r < 8; ++r) {
                Mout[(size_t)(e0 + rbase + r) * OUTD + off]      = (_Float16)(c0a[r] + bv);
                Mout[(size_t)(e0 + 16 + rbase + r) * OUTD + off] = (_Float16)(c1a[r] + bv);
            }
        }
    } else {
        // n = wave*512 + t*16 + col ; i = n>>6 = 8*wave + (t>>2) ; j = n&63
        float m0[8] = {}, m1[8] = {};
        for (int t = 0; t < 32; ++t) {
            int ntile = wave * 32 + t;
            if (t < 31) {
                #pragma unroll
                for (int kt = 0; kt < 4; ++kt)
                    __builtin_prefetch(W7 + ((ntile + 1) * 4 + kt) * 512 + lane * 16, 0, 3);
            }
            v8f c0a = {}, c1a = {};
            #pragma unroll
            for (int kt = 0; kt < 4; ++kt) {
                v16h b = load_b_frag(W7 + (ntile * 4 + kt) * 512, lane);
                c0a = wmma16(a4[0][kt], b, c0a);
                c1a = wmma16(a4[1][kt], b, c1a);
            }
            float bv = b7[ntile * 16 + col];
            int  j   = ((t & 3) << 4) + col;
            #pragma unroll
            for (int r = 0; r < 8; ++r) {
                m0[r] += (c0a[r] + bv) * xsrcS[rbase + r][j];
                m1[r] += (c1a[r] + bv) * xsrcS[16 + rbase + r][j];
            }
            if ((t & 3) == 3) {
                #pragma unroll
                for (int m = 1; m < 16; m <<= 1)
                    #pragma unroll
                    for (int r = 0; r < 8; ++r) {
                        m0[r] += __shfl_xor(m0[r], m, 32);
                        m1[r] += __shfl_xor(m1[r], m, 32);
                    }
                if (col == 0) {            // lanes 0,16; column i exclusive to wave
                    int i = wave * 8 + (t >> 2);
                    #pragma unroll
                    for (int r = 0; r < 8; ++r) {
                        lmsg[rbase + r][i]      += m0[r];
                        lmsg[16 + rbase + r][i] += m1[r];
                    }
                }
                #pragma unroll
                for (int r = 0; r < 8; ++r) { m0[r] = 0.f; m1[r] = 0.f; }
            }
        }
        __syncthreads();
        #pragma unroll
        for (int q = 0; q < 8; ++q) {
            int id = tid * 8 + q;
            int e  = id >> 6, i = id & 63;
            atomicAdd(&num[(size_t)edstS[e] * DIMX + i], ewS[e] * lmsg[e][i]);
        }
    }
}

// ---------------------------------------------------------------------------
// loss_orth: per pair p, P = M[p] @ M[p+EH]; acc += sqrt(sum((P-I)^2)).
// One wave per pair: 32 WMMAs. B half of M is stored transposed (K-contiguous),
// so both operand fragment loads are contiguous. Uses
// sum((P-I)^2) = sum(P^2) - 2*trace(P) + 64 to avoid per-element compares.
// ---------------------------------------------------------------------------
__global__ void __launch_bounds__(256) orth_kernel(
    const _Float16* __restrict__ M0, int EHn, float* __restrict__ acc)
{
    const int lane = threadIdx.x & 31;
    const int wave = threadIdx.x >> 5;
    const int p    = blockIdx.x * 8 + wave;
    if (p >= EHn) return;
    const _Float16* A  = M0 + (size_t)p * OUTD;            // row-major
    const _Float16* Bt = M0 + (size_t)(p + EHn) * OUTD;    // transposed storage
    const int col   = lane & 15;
    const int rbase = (lane & 16) ? 8 : 0;

    v16h af[4][2];
    #pragma unroll
    for (int mi = 0; mi < 4; ++mi)
        #pragma unroll
        for (int kt = 0; kt < 2; ++kt)
            af[mi][kt] = load_a_frag(A + mi * 16 * 64, 64, lane, kt);

    float ssq = 0.f, tr = 0.f;
    const int rr = col - rbase;            // diagonal element index (if 0..7)
    for (int ni = 0; ni < 4; ++ni) {
        v16h bf[2];
        #pragma unroll
        for (int kt = 0; kt < 2; ++kt)
            bf[kt] = load_b_frag_kc(Bt, lane, kt, ni * 16);
        #pragma unroll
        for (int mi = 0; mi < 4; ++mi) {
            v8f c = {};
            c = wmma16(af[mi][0], bf[0], c);
            c = wmma16(af[mi][1], bf[1], c);
            #pragma unroll
            for (int r = 0; r < 8; ++r) ssq += c[r] * c[r];
            if (mi == ni && rr >= 0 && rr < 8) tr += c[rr];
        }
    }
    #pragma unroll
    for (int m = 1; m < 32; m <<= 1) {
        ssq += __shfl_xor(ssq, m, 32);
        tr  += __shfl_xor(tr,  m, 32);
    }
    if (lane == 0) atomicAdd(acc, sqrtf(ssq - 2.f * tr + 64.f));
}

// xnew = cnt>0 ? num/cnt : 0
__global__ void __launch_bounds__(256) norm_kernel(
    const float* __restrict__ num, const float* __restrict__ cnt,
    float* __restrict__ xout, int n)
{
    int i = blockIdx.x * 256 + threadIdx.x;
    if (i >= n) return;
    float c = cnt[i >> 6];
    xout[i] = (c > 0.f) ? num[i] / c : 0.f;
}

// acc[1] += sum((a-b)^2)
__global__ void __launch_bounds__(256) smap_kernel(
    const float* __restrict__ a, const float* __restrict__ b, int n,
    float* __restrict__ acc)
{
    int   i = blockIdx.x * 256 + threadIdx.x;
    float d = (i < n) ? (a[i] - b[i]) : 0.f;
    d *= d;
    #pragma unroll
    for (int m = 1; m < 32; m <<= 1) d += __shfl_xor(d, m, 32);
    if ((threadIdx.x & 31) == 0) atomicAdd(acc + 1, d);
}

// classifier: per idvert row, 16 lanes = 16 classes; log_softmax, KL term, argmax
__global__ void __launch_bounds__(256) cls_kernel(
    const float* __restrict__ xm, const int* __restrict__ idvert,
    const float* __restrict__ ylprob, const int* __restrict__ ylabel,
    const float* __restrict__ Wc, const float* __restrict__ bc,
    int nidv, float* __restrict__ acc)
{
    int tid = threadIdx.x;
    int c   = tid & 15;
    int row = blockIdx.x * 16 + (tid >> 4);
    if (row >= nidv) return;
    int v = idvert[row];
    float z = bc[c];
    #pragma unroll
    for (int j = 0; j < 64; ++j) z += xm[(size_t)v * 64 + j] * Wc[j * 16 + c];
    float mx = z;
    #pragma unroll
    for (int m = 1; m < 16; m <<= 1) mx = fmaxf(mx, __shfl_xor(mx, m, 32));
    float s = expf(z - mx);
    #pragma unroll
    for (int m = 1; m < 16; m <<= 1) s += __shfl_xor(s, m, 32);
    float gl  = z - (logf(s) + mx);
    float ylv = ylprob[(size_t)v * 16 + c];
    float t   = expf(ylv) * (ylv - gl);
    #pragma unroll
    for (int m = 1; m < 16; m <<= 1) t += __shfl_xor(t, m, 32);
    float bz = z; int bi = c;                       // argmax, first-index tie-break
    #pragma unroll
    for (int m = 1; m < 16; m <<= 1) {
        float oz = __shfl_xor(bz, m, 32);
        int   oi = __shfl_xor(bi, m, 32);
        if (oz > bz || (oz == bz && oi < bi)) { bz = oz; bi = oi; }
    }
    if (c == 0) {
        atomicAdd(acc + 2, t);
        atomicAdd(acc + 3, (bi == ylabel[v]) ? 1.f : 0.f);
    }
}

__global__ void fin_kernel(const float* __restrict__ acc, float* __restrict__ out,
                           int EHn, int nv, int nidv)
{
    out[0] = acc[0] / (float)EHn;   // loss_orth
    out[1] = 0.f;                   // loss_cons: Mf recomputed from identical inputs
    out[2] = acc[1] / (float)nv;    // mean((xm-x)^2)*64 = sum/(nv*64)*64
    out[3] = acc[2] / (float)nidv;  // loss_lbpr
    out[4] = acc[3] / (float)nidv;  // loss_accs
}

// ---------------------------------------------------------------------------
static inline size_t align256(size_t x) { return (x + 255) & ~(size_t)255; }

extern "C" void kernel_launch(void* const* d_in, const int* in_sizes, int n_in,
                              void* d_out, int out_size, void* d_ws, size_t ws_size,
                              hipStream_t stream)
{
    const float* xembed = (const float*)d_in[0];
    const int*   ylabel = (const int*)  d_in[1];
    const float* ylprob = (const float*)d_in[2];
    const int*   esrc   = (const int*)  d_in[3];
    const int*   edst   = (const int*)  d_in[4];
    const float* ew     = (const float*)d_in[5];
    const int*   idvert = (const int*)  d_in[6];
    const float* W[8]; const float* b[8];
    for (int i = 0; i < 8; ++i) {
        W[i] = (const float*)d_in[7 + 2 * i];
        b[i] = (const float*)d_in[8 + 2 * i];
    }
    const float* Wc = (const float*)d_in[23];
    const float* bc = (const float*)d_in[24];

    const int E    = in_sizes[3];
    const int EHn  = E / 2;
    const int NV   = in_sizes[0] / DIMX;
    const int NIDV = in_sizes[6];

    // workspace carve-up
    char* w = (char*)d_ws;
    _Float16* Wp  = (_Float16*)w; w += align256((size_t)WP_TOTAL * 2);
    float*    Bb  = (float*)w;    w += align256((size_t)BB_TOTAL * 4);
    _Float16* M0  = (_Float16*)w; w += align256((size_t)E * OUTD * 2);   // 128 MB
    float*    xa  = (float*)w;    w += align256((size_t)NV * DIMX * 4);
    float*    xb  = (float*)w;    w += align256((size_t)NV * DIMX * 4);
    float*    num = (float*)w;    w += align256((size_t)NV * DIMX * 4);
    float*    cnt = (float*)w;    w += align256((size_t)NV * 4);
    float*    acc = (float*)w;    w += 256;

    hipMemsetAsync(acc, 0, 4 * sizeof(float), stream);
    hipMemsetAsync(cnt, 0, (size_t)NV * sizeof(float), stream);

    pack_w_kernel<<<(WP_TOTAL + 255) / 256, 256, 0, stream>>>(
        W[0], W[1], W[2], W[3], W[4], W[5], W[6], W[7], Wp);
    pack_b_kernel<<<(BB_TOTAL + 255) / 256, 256, 0, stream>>>(
        b[0], b[1], b[2], b[3], b[4], b[5], b[6], b[7], Bb);
    cnt_kernel<<<(E + 255) / 256, 256, 0, stream>>>(edst, ew, cnt, E);

    // M0 = MLP(xembed) materialized once (for loss_orth; 2nd half transposed)
    mlp_edge_kernel<false><<<E / EPB, 256, 0, stream>>>(
        xembed, esrc, edst, ew, Wp, Bb, M0, nullptr, EHn);
    orth_kernel<<<(EHn + 7) / 8, 256, 0, stream>>>(M0, EHn, acc);

    // 3 conv iterations, matvec fused into the final-layer WMMA epilogue
    const float* xin = xembed;
    float* bufs[3] = {xa, xb, xa};
    for (int it = 0; it < 3; ++it) {
        hipMemsetAsync(num, 0, (size_t)NV * DIMX * sizeof(float), stream);
        mlp_edge_kernel<true><<<E / EPB, 256, 0, stream>>>(
            xin, esrc, edst, ew, Wp, Bb, nullptr, num, EHn);
        norm_kernel<<<(NV * DIMX + 255) / 256, 256, 0, stream>>>(
            num, cnt, bufs[it], NV * DIMX);
        xin = bufs[it];
    }

    smap_kernel<<<(NV * DIMX + 255) / 256, 256, 0, stream>>>(
        xin, xembed, NV * DIMX, acc);
    cls_kernel<<<(NIDV + 15) / 16, 256, 0, stream>>>(
        xin, idvert, ylprob, ylabel, Wc, bc, NIDV, acc);
    fin_kernel<<<1, 1, 0, stream>>>(acc, (float*)d_out, EHn, NV, NIDV);
}
